// MiniMoE_33157147525722
// MI455X (gfx1250) — matrix operand order repaired and checked by reading, not compiled, not run on hardware
//
#include <hip/hip_runtime.h>
#include <hip/hip_bf16.h>
#include <cmath>
#include <cstdint>

#define TKN   8192
#define DDIM  1024
#define FDIM  4096
#define NEXP  4

typedef __attribute__((ext_vector_type(16))) __bf16 v16bf;
typedef __attribute__((ext_vector_type(8)))  __bf16 v8bf;
typedef __attribute__((ext_vector_type(8)))  float  v8f;

union Frag { v16bf v; v8bf h[2]; };

constexpr int BM = 128;   // token rows per block
constexpr int BN = 64;    // output cols per block
constexpr int KC = 32;    // k-chunk (WMMA bf16 K)
constexpr int LSTR = 40;  // LDS row stride in bf16 (80B, conflict-friendly)

// ---- CDNA5 async memory->LDS DMA (ASYNCcnt path), 16B per lane ----
__device__ __forceinline__ void async_ld_b128(uint32_t lds_byte_off, const void* gptr) {
  asm volatile("global_load_async_to_lds_b128 %0, %1, off"
               :: "v"(lds_byte_off), "v"(gptr) : "memory");
}
__device__ __forceinline__ void wait_async0() {
  asm volatile("s_wait_asynccnt 0x0" ::: "memory");
}
__device__ __forceinline__ uint32_t lds_off(const void* p) {
  return (uint32_t)(uintptr_t)p;   // generic LDS address: offset lives in addr[31:0]
}

// ---------------- utility kernels ----------------

__global__ void zero_kernel(float* __restrict__ out, int* __restrict__ counts,
                            int* __restrict__ tok, float* __restrict__ gat) {
  int idx = blockIdx.x * blockDim.x + threadIdx.x;
  if (idx < TKN * DDIM) out[idx] = 0.0f;
  if (idx < NEXP * TKN) { tok[idx] = 0; gat[idx] = 0.0f; }
  if (idx < NEXP) counts[idx] = 0;
}

__global__ void cvt_kernel(const float* __restrict__ in, __bf16* __restrict__ out, int n) {
  int idx = blockIdx.x * blockDim.x + threadIdx.x;
  if (idx < n) out[idx] = (__bf16)in[idx];
}

// [R][C] fp32 -> [C][R] bf16, batched over blockIdx.z
__global__ void __launch_bounds__(256)
tcvt_kernel(const float* __restrict__ in, __bf16* __restrict__ out, int R, int C) {
  __shared__ float tile[32][33];
  const float* src = in + (size_t)blockIdx.z * R * C;
  __bf16* dst = out + (size_t)blockIdx.z * R * C;
  int c0 = blockIdx.x * 32, r0 = blockIdx.y * 32;
  int tx = threadIdx.x, ty = threadIdx.y;
  #pragma unroll
  for (int j = 0; j < 32; j += 8)
    tile[ty + j][tx] = src[(size_t)(r0 + ty + j) * C + c0 + tx];
  __syncthreads();
  #pragma unroll
  for (int j = 0; j < 32; j += 8)
    dst[(size_t)(c0 + ty + j) * R + r0 + tx] = (__bf16)tile[tx][ty + j];
}

// ---------------- router: one wave per token ----------------

__global__ void __launch_bounds__(256)
router_kernel(const float* __restrict__ x, const float* __restrict__ rw,
              const float* __restrict__ rb, int* __restrict__ counts,
              int* __restrict__ tok, float* __restrict__ gat) {
  const int wave = threadIdx.x >> 5, lane = threadIdx.x & 31;
  const int t = blockIdx.x * 8 + wave;
  if (t >= TKN) return;
  float a0 = 0.f, a1 = 0.f, a2 = 0.f, a3 = 0.f;
  const float* xr = x + (size_t)t * DDIM;
  for (int d = lane; d < DDIM; d += 32) {
    float xv = xr[d];
    float4 w = *(const float4*)(rw + d * 4);
    a0 += xv * w.x; a1 += xv * w.y; a2 += xv * w.z; a3 += xv * w.w;
  }
  #pragma unroll
  for (int off = 16; off; off >>= 1) {
    a0 += __shfl_down(a0, off);
    a1 += __shfl_down(a1, off);
    a2 += __shfl_down(a2, off);
    a3 += __shfl_down(a3, off);
  }
  if (lane == 0) {
    float l[NEXP] = {a0 + rb[0], a1 + rb[1], a2 + rb[2], a3 + rb[3]};
    float mx = fmaxf(fmaxf(l[0], l[1]), fmaxf(l[2], l[3]));
    float p[NEXP], s = 0.f;
    #pragma unroll
    for (int i = 0; i < NEXP; ++i) { p[i] = expf(l[i] - mx); s += p[i]; }
    #pragma unroll
    for (int i = 0; i < NEXP; ++i) p[i] /= s;
    int i0 = 0;
    #pragma unroll
    for (int i = 1; i < NEXP; ++i) if (p[i] > p[i0]) i0 = i;
    int i1 = (i0 == 0) ? 1 : 0;
    #pragma unroll
    for (int i = 0; i < NEXP; ++i) if (i != i0 && p[i] > p[i1]) i1 = i;
    float den = p[i0] + p[i1];
    int s0 = atomicAdd(&counts[i0], 1);
    tok[i0 * TKN + s0] = t; gat[i0 * TKN + s0] = p[i0] / den;
    int s1 = atomicAdd(&counts[i1], 1);
    tok[i1 * TKN + s1] = t; gat[i1 * TKN + s1] = p[i1] / den;
  }
}

// ---------------- GEMM1: h = gate * silu(x@w1) * (x@w3), routed ----------------

__global__ void __launch_bounds__(256)
gemm1_kernel(const __bf16* __restrict__ xbf,
             const __bf16* __restrict__ w1T, const __bf16* __restrict__ w3T,
             const int* __restrict__ tok, const float* __restrict__ gat,
             const int* __restrict__ counts, __bf16* __restrict__ hout) {
  __shared__ __bf16 As[2][BM * LSTR];
  __shared__ __bf16 Bs[2][2][BN * LSTR];
  __shared__ __bf16 Ep[8][32 * LSTR];
  __shared__ int    tokS[BM];

  const int e = blockIdx.z;
  const int m0 = blockIdx.y * BM;
  if (m0 >= counts[e]) return;
  const int f0 = blockIdx.x * BN;
  const int tid = threadIdx.x;
  const int wave = tid >> 5, lane = tid & 31;
  const int lr = lane & 15, lh = lane >> 4;
  const int mw = wave >> 1, nw = wave & 1;

  if (tid < BM) tokS[tid] = tok[e * TKN + m0 + tid];
  __syncthreads();

  const __bf16* w1p = w1T + ((size_t)e * FDIM + f0) * DDIM;
  const __bf16* w3p = w3T + ((size_t)e * FDIM + f0) * DDIM;

  // per-thread staging sources (hoisted out of the k-loop)
  const int r4 = tid >> 2, seg = tid & 3;            // r4: 0..63, seg: 16B segment
  const __bf16* aSrc0 = xbf + (size_t)tokS[r4] * DDIM + seg * 8;
  const __bf16* aSrc1 = xbf + (size_t)tokS[r4 + 64] * DDIM + seg * 8;
  const __bf16* bSrc0 = w1p + (size_t)r4 * DDIM + seg * 8;
  const __bf16* bSrc1 = w3p + (size_t)r4 * DDIM + seg * 8;

  const uint32_t ldsA  = lds_off(&As[0][0]);
  const uint32_t ldsB  = lds_off(&Bs[0][0][0]);
  const uint32_t offA0 = 2u * (r4 * LSTR + seg * 8);
  const uint32_t offA1 = 2u * ((r4 + 64) * LSTR + seg * 8);
  const uint32_t offB0 = 2u * (r4 * LSTR + seg * 8);
  const uint32_t offB1 = offB0 + 2u * BN * LSTR;
  const uint32_t bufA  = 2u * BM * LSTR;
  const uint32_t bufB  = 2u * 2 * BN * LSTR;

  v8f acc[2][2][2] = {};

  auto stage = [&](int buf, int k) {
    async_ld_b128(ldsA + buf * bufA + offA0, aSrc0 + k);
    async_ld_b128(ldsA + buf * bufA + offA1, aSrc1 + k);
    async_ld_b128(ldsB + buf * bufB + offB0, bSrc0 + k);
    async_ld_b128(ldsB + buf * bufB + offB1, bSrc1 + k);
  };

  stage(0, 0);
  wait_async0();
  __syncthreads();

  const int NSTEP = DDIM / KC;
  for (int ks = 0; ks < NSTEP; ++ks) {
    const int cur = ks & 1;
    if (ks + 1 < NSTEP) stage(cur ^ 1, (ks + 1) * KC);

    Frag a[2], b[2][2];
    #pragma unroll
    for (int mt = 0; mt < 2; ++mt) {           // A: lane half picks K {0..7,16..23}/{8..15,24..31}
      const __bf16* base = &As[cur][(mw * 32 + mt * 16 + lr) * LSTR];
      a[mt].h[0] = *(const v8bf*)(base + lh * 8);
      a[mt].h[1] = *(const v8bf*)(base + 16 + lh * 8);
    }
    #pragma unroll
    for (int m = 0; m < 2; ++m)
      #pragma unroll
      for (int nt = 0; nt < 2; ++nt) {         // B: lane half picks K {0..7,16..23}/{8..15,24..31}
        const __bf16* base = &Bs[cur][m][(nw * 32 + nt * 16 + lr) * LSTR + lh * 8];
        b[m][nt].h[0] = *(const v8bf*)(base);
        b[m][nt].h[1] = *(const v8bf*)(base + 16);
      }
    #pragma unroll
    for (int m = 0; m < 2; ++m)
      #pragma unroll
      for (int mt = 0; mt < 2; ++mt)
        #pragma unroll
        for (int nt = 0; nt < 2; ++nt)
          acc[m][mt][nt] = __builtin_amdgcn_wmma_f32_16x16x32_bf16(
              false, a[mt].v, false, b[m][nt].v, (short)0, acc[m][mt][nt], false, false);

    wait_async0();        // my next-buffer fills have landed in LDS
    __syncthreads();      // everyone's fills visible; reads of cur done
  }

  // epilogue: gate * silu(h1) * h3 -> bf16, stage in LDS, coalesced store
  #pragma unroll
  for (int mt = 0; mt < 2; ++mt)
    #pragma unroll
    for (int v = 0; v < 8; ++v) {
      const int rloc = mt * 16 + lh * 8 + v;
      const float g = gat[e * TKN + m0 + mw * 32 + rloc];
      #pragma unroll
      for (int nt = 0; nt < 2; ++nt) {
        float h1 = acc[0][mt][nt][v];
        float h3 = acc[1][mt][nt][v];
        float sg = h1 / (1.0f + expf(-h1));
        Ep[wave][rloc * LSTR + nt * 16 + lr] = (__bf16)(g * sg * h3);
      }
    }
  __syncthreads();
  #pragma unroll
  for (int p = 0; p < 4; ++p) {
    int idx = p * 32 + lane;
    int r = idx >> 2, sg2 = idx & 3;
    v8bf val = *(const v8bf*)&Ep[wave][r * LSTR + sg2 * 8];
    *(v8bf*)&hout[((size_t)(e * TKN + m0 + mw * 32 + r)) * FDIM + f0 + nw * 32 + sg2 * 8] = val;
  }
}

// ---------------- GEMM2: out[token] += h @ w2, routed ----------------

__global__ void __launch_bounds__(256)
gemm2_kernel(const __bf16* __restrict__ hin, const __bf16* __restrict__ w2T,
             const int* __restrict__ tok, const int* __restrict__ counts,
             float* __restrict__ out) {
  __shared__ __bf16 As[2][BM * LSTR];
  __shared__ __bf16 Bs[2][BN * LSTR];
  __shared__ int    tokS[BM];

  const int e = blockIdx.z;
  const int m0 = blockIdx.y * BM;
  if (m0 >= counts[e]) return;
  const int d0 = blockIdx.x * BN;
  const int tid = threadIdx.x;
  const int wave = tid >> 5, lane = tid & 31;
  const int lr = lane & 15, lh = lane >> 4;
  const int mw = wave >> 1, nw = wave & 1;

  if (tid < BM) tokS[tid] = tok[e * TKN + m0 + tid];

  const __bf16* ap = hin + ((size_t)(e * TKN + m0)) * FDIM;
  const __bf16* bp = w2T + ((size_t)(e * DDIM + d0)) * FDIM;

  const int r4 = tid >> 2, seg = tid & 3;
  const __bf16* aSrc0 = ap + (size_t)r4 * FDIM + seg * 8;
  const __bf16* aSrc1 = ap + (size_t)(r4 + 64) * FDIM + seg * 8;
  const __bf16* bSrc  = bp + (size_t)r4 * FDIM + seg * 8;

  const uint32_t ldsA  = lds_off(&As[0][0]);
  const uint32_t ldsB  = lds_off(&Bs[0][0]);
  const uint32_t offA0 = 2u * (r4 * LSTR + seg * 8);
  const uint32_t offA1 = 2u * ((r4 + 64) * LSTR + seg * 8);
  const uint32_t bufA  = 2u * BM * LSTR;
  const uint32_t bufB  = 2u * BN * LSTR;

  v8f acc[2][2] = {};

  auto stage = [&](int buf, int k) {
    async_ld_b128(ldsA + buf * bufA + offA0, aSrc0 + k);
    async_ld_b128(ldsA + buf * bufA + offA1, aSrc1 + k);
    async_ld_b128(ldsB + buf * bufB + offA0, bSrc + k);
  };

  stage(0, 0);
  wait_async0();
  __syncthreads();

  const int NSTEP = FDIM / KC;
  for (int ks = 0; ks < NSTEP; ++ks) {
    const int cur = ks & 1;
    if (ks + 1 < NSTEP) stage(cur ^ 1, (ks + 1) * KC);

    Frag a[2], b[2];
    #pragma unroll
    for (int mt = 0; mt < 2; ++mt) {
      const __bf16* base = &As[cur][(mw * 32 + mt * 16 + lr) * LSTR];
      a[mt].h[0] = *(const v8bf*)(base + lh * 8);
      a[mt].h[1] = *(const v8bf*)(base + 16 + lh * 8);
    }
    #pragma unroll
    for (int nt = 0; nt < 2; ++nt) {
      const __bf16* base = &Bs[cur][(nw * 32 + nt * 16 + lr) * LSTR + lh * 8];
      b[nt].h[0] = *(const v8bf*)(base);
      b[nt].h[1] = *(const v8bf*)(base + 16);
    }
    #pragma unroll
    for (int mt = 0; mt < 2; ++mt)
      #pragma unroll
      for (int nt = 0; nt < 2; ++nt)
        acc[mt][nt] = __builtin_amdgcn_wmma_f32_16x16x32_bf16(
            false, a[mt].v, false, b[nt].v, (short)0, acc[mt][nt], false, false);

    wait_async0();
    __syncthreads();
  }

  // epilogue: scatter-accumulate into out (gate already folded into h)
  #pragma unroll
  for (int mt = 0; mt < 2; ++mt)
    #pragma unroll
    for (int v = 0; v < 8; ++v) {
      const int rloc = mt * 16 + lh * 8 + v;
      const int t = tokS[mw * 32 + rloc];
      #pragma unroll
      for (int nt = 0; nt < 2; ++nt)
        unsafeAtomicAdd(out + (size_t)t * DDIM + d0 + nw * 32 + nt * 16 + lr,
                        acc[mt][nt][v]);
    }
}

// ---------------- launch ----------------

extern "C" void kernel_launch(void* const* d_in, const int* in_sizes, int n_in,
                              void* d_out, int out_size, void* d_ws, size_t ws_size,
                              hipStream_t stream) {
  (void)in_sizes; (void)n_in; (void)out_size; (void)ws_size;
  const float* x  = (const float*)d_in[0];
  const float* rw = (const float*)d_in[1];
  const float* rb = (const float*)d_in[2];
  const float* w1 = (const float*)d_in[3];
  const float* w3 = (const float*)d_in[4];
  const float* w2 = (const float*)d_in[5];
  float* out = (float*)d_out;

  char* ws = (char*)d_ws;
  size_t o = 0;
  __bf16* xbf = (__bf16*)(ws + o); o += (size_t)TKN * DDIM * 2;
  __bf16* w1T = (__bf16*)(ws + o); o += (size_t)NEXP * DDIM * FDIM * 2;
  __bf16* w3T = (__bf16*)(ws + o); o += (size_t)NEXP * DDIM * FDIM * 2;
  __bf16* w2T = (__bf16*)(ws + o); o += (size_t)NEXP * DDIM * FDIM * 2;
  __bf16* hbuf = (__bf16*)(ws + o); o += (size_t)NEXP * TKN * FDIM * 2;
  int*   tok    = (int*)(ws + o);  o += (size_t)NEXP * TKN * 4;
  float* gat    = (float*)(ws + o); o += (size_t)NEXP * TKN * 4;
  int*   counts = (int*)(ws + o);  o += 256;

  const int nx = TKN * DDIM;
  zero_kernel<<<(nx + 255) / 256, 256, 0, stream>>>(out, counts, tok, gat);
  cvt_kernel<<<(nx + 255) / 256, 256, 0, stream>>>(x, xbf, nx);

  dim3 tb(32, 8);
  tcvt_kernel<<<dim3(FDIM / 32, DDIM / 32, NEXP), tb, 0, stream>>>(w1, w1T, DDIM, FDIM);
  tcvt_kernel<<<dim3(FDIM / 32, DDIM / 32, NEXP), tb, 0, stream>>>(w3, w3T, DDIM, FDIM);
  tcvt_kernel<<<dim3(DDIM / 32, FDIM / 32, NEXP), tb, 0, stream>>>(w2, w2T, FDIM, DDIM);

  router_kernel<<<TKN / 8, 256, 0, stream>>>(x, rw, rb, counts, tok, gat);

  gemm1_kernel<<<dim3(FDIM / BN, TKN / BM, NEXP), 256, 0, stream>>>(
      xbf, w1T, w3T, tok, gat, counts, hbuf);
  gemm2_kernel<<<dim3(DDIM / BN, TKN / BM, NEXP), 256, 0, stream>>>(
      hbuf, w2T, tok, counts, out);
}
